// LSTM_62912680952312
// MI455X (gfx1250) — matrix operand order repaired and checked by reading, not compiled, not run on hardware
//
#include <hip/hip_runtime.h>

// Problem constants (match reference)
#define Bsz   512
#define Tsz   512
#define Hsz   100
#define Lsz   4
#define G4    400      // 4*H gates
#define HP    128      // hidden padded to 128 for interleaved gate layout
#define NP    512      // padded N = 4*HP -> 32 WMMA N-tiles, 4 per wave
#define KPAD  256      // K layout: [h(0..127) | x_t | 1.0 (bias) | 0-pad]
#define XOFF  128      // input column offset inside A tile
#define BT    16       // batch tile per workgroup (one WMMA M-tile)
#define BLK   256      // 8 wave32 waves

typedef __bf16 v16bf __attribute__((ext_vector_type(16)));
typedef float  v8f   __attribute__((ext_vector_type(8)));

union FragAB { v16bf v; unsigned int u[8]; };
union FragC  { v8f   v; float        f[8]; };

__device__ __forceinline__ unsigned short f2bf(float f) {
  unsigned int u = __float_as_uint(f);
  u += 0x7FFFu + ((u >> 16) & 1u);          // round-to-nearest-even
  return (unsigned short)(u >> 16);
}
// sigmoid via v_exp + v_rcp (no IEEE div expansion)
__device__ __forceinline__ float sigmoidf_(float x) {
  return __builtin_amdgcn_rcpf(1.0f + __expf(-x));
}
// tanh(x) = 2*sigmoid(2x) - 1
__device__ __forceinline__ float tanh_fast(float x) {
  return __builtin_fmaf(2.0f, sigmoidf_(2.0f * x), -1.0f);
}
__device__ __forceinline__ float quad_bcast(float v, int srcLane) {
  return __int_as_float(__builtin_amdgcn_ds_bpermute(srcLane << 2, __float_as_int(v)));
}

// ---------------------------------------------------------------------------
// Pack weights bf16, gate-interleaved (n = 4*h + g, h padded to 128), K padded
// to 256 with layout [W_hh(128) | W_ih(in_dim) | bias | 0]. n-major so the
// B-fragment dwords (packed K-pairs) are contiguous.
// ---------------------------------------------------------------------------
__global__ void lstm_pack_weights(const float* __restrict__ W_ih0,
                                  const float* __restrict__ W_ih_rest,
                                  const float* __restrict__ W_hh,
                                  const float* __restrict__ b_ih,
                                  const float* __restrict__ b_hh,
                                  unsigned short* __restrict__ wbuf) {
  int idx = blockIdx.x * blockDim.x + threadIdx.x;
  const int total = Lsz * NP * KPAD;
  if (idx >= total) return;
  int k = idx % KPAD;
  int n = (idx / KPAD) % NP;
  int l = idx / (KPAD * NP);
  int g = n & 3, h = n >> 2;
  int in_dim = (l == 0) ? 1 : Hsz;
  float v = 0.0f;
  if (h < Hsz) {
    int j = g * Hsz + h;                       // PyTorch gate row (i,f,g,o)
    if (k < Hsz)
      v = W_hh[((size_t)l * G4 + j) * Hsz + k];
    else if (k >= XOFF && k < XOFF + in_dim)
      v = (l == 0) ? W_ih0[j]
                   : W_ih_rest[((size_t)(l - 1) * G4 + j) * Hsz + (k - XOFF)];
    else if (k == XOFF + in_dim)               // bias row (A has 1.0 here)
      v = b_ih[l * G4 + j] + b_hh[l * G4 + j];
  }
  wbuf[idx] = f2bf(v);
}

// ---------------------------------------------------------------------------
// Guard-free fused GEMM: each wave owns 4 consecutive N-tiles, fully unrolled.
// A from LDS (ds_load_b128 pairs), B from L2-resident packed weights.
// ---------------------------------------------------------------------------
template <int KS>
__device__ __forceinline__ void gemm_step(const unsigned short* __restrict__ ashp,
                                          const unsigned short* __restrict__ wL,
                                          int wave, int lane, FragC acc[4]) {
  const int mrow = lane & 15;
  const int half = lane >> 4;
  const unsigned int* aptr = (const unsigned int*)ashp;
  const unsigned int* wptr = (const unsigned int*)wL;
#pragma unroll
  for (int ks = 0; ks < KS; ++ks) {
    FragAB a;
    const int ka = ks * 32 + 8 * half;
#pragma unroll
    for (int v = 0; v < 4; ++v) a.u[v]     = aptr[(mrow * KPAD + ka + 2 * v) >> 1];
#pragma unroll
    for (int v = 0; v < 4; ++v) a.u[4 + v] = aptr[(mrow * KPAD + ka + 16 + 2 * v) >> 1];
    const int kb = ks * 32 + 16 * half;
#pragma unroll
    for (int i = 0; i < 4; ++i) {
      const int n = (wave * 4 + i) * 16 + mrow;
      FragAB b;
#pragma unroll
      for (int v = 0; v < 8; ++v)
        b.u[v] = wptr[(((size_t)n * KPAD) + kb - 8 * half + 2 * (v & 3) + 16 * (v >> 2)) >> 1];
      acc[i].v = __builtin_amdgcn_wmma_f32_16x16x32_bf16(
          false, a.v, false, b.v, (short)0, acc[i].v, false, false);
    }
  }
}

// ---------------------------------------------------------------------------
// Persistent LSTM: each workgroup owns 16 batch rows, runs all layers/steps.
// Cell update is register-resident: each lane activates its OWN gate once
// (branchless sigmoid/tanh unification), then 4x ds_bpermute quad-broadcast
// of the activated values; c-state lives in VGPRs.
// ---------------------------------------------------------------------------
__global__ __launch_bounds__(BLK) void lstm_persist(
    const float* __restrict__ x,
    const unsigned short* __restrict__ wbuf,
    const float* __restrict__ W_out, const float* __restrict__ b_out,
    const float* __restrict__ dmask,
    unsigned short* __restrict__ actA, unsigned short* __restrict__ actB,
    float* __restrict__ out) {
  __shared__ unsigned short ash[BT * KPAD];  // bf16 A-tile [h | x_t | 1 | 0]
  __shared__ float oacc[BT];                 // final projection partials

  const int tid  = threadIdx.x;
  const int lane = tid & 31;
  const int wave = __builtin_amdgcn_readfirstlane(tid >> 5);
  const int half = lane >> 4;
  const int quad = (lane & 15) >> 2;         // hidden index within N-tile
  const int g    = lane & 3;                 // gate id within quad (i,f,g,o)
  const int qbase = lane & ~3;
  const int bbase = blockIdx.x * BT;

  // per-lane gate nonlinearity: y = sigmoid(s*x); act = fma(aa, y, bb)
  // g==2 ("g" gate) -> tanh(x) = 2*sigmoid(2x)-1 ; others -> sigmoid(x)
  const float s_  = (g == 2) ? 2.0f : 1.0f;
  const float aa_ = (g == 2) ? 2.0f : 1.0f;
  const float bb_ = (g == 2) ? -1.0f : 0.0f;

  if (tid < BT) oacc[tid] = 0.0f;

  for (int layer = 0; layer < Lsz; ++layer) {
    const int in_dim = (layer == 0) ? 1 : Hsz;
    const unsigned short* wL = wbuf + (size_t)layer * NP * KPAD;
    const unsigned short* actIn  = (layer == 2) ? actB : actA;
    unsigned short*       actOut = (layer == 1) ? actB : actA;

    for (int i = tid; i < BT * KPAD; i += BLK) ash[i] = 0;     // h0 = 0 + 0-pad
    float cst[4][8];                                            // c0 = 0
#pragma unroll
    for (int i = 0; i < 4; ++i)
#pragma unroll
      for (int r = 0; r < 8; ++r) cst[i][r] = 0.0f;
    __syncthreads();
    if (tid < BT) ash[tid * KPAD + XOFF + in_dim] = 0x3F80;    // bias 1.0 col
    __syncthreads();

    for (int t = 0; t < Tsz; ++t) {
      // --- stage this step's input into A-tile cols [XOFF, XOFF+in_dim) ---
      if (layer == 0) {
        if (tid < BT)
          ash[tid * KPAD + XOFF] = f2bf(x[(size_t)(bbase + tid) * Tsz + t]);
      } else {
        for (int i = tid; i < BT * Hsz; i += BLK) {
          int b = i / Hsz, h = i - b * Hsz;
          ash[b * KPAD + XOFF + h] = actIn[((size_t)(bbase + b) * Tsz + t) * Hsz + h];
        }
        if (t + 1 < Tsz && tid < BT)  // prefetch next step's input row
          __builtin_prefetch(&actIn[((size_t)(bbase + tid) * Tsz + t + 1) * Hsz], 0, 1);
      }
      __syncthreads();

      // --- fused [h | x_t | 1] @ Wc^T via WMMA bf16 (bias included) ---
      FragC acc[4];
#pragma unroll
      for (int i = 0; i < 4; ++i)
#pragma unroll
        for (int r = 0; r < 8; ++r) acc[i].f[r] = 0.0f;

      if (layer == 0) gemm_step<5>(ash, wL, wave, lane, acc);   // K = 130
      else            gemm_step<8>(ash, wL, wave, lane, acc);   // K = 229

      // --- register-resident LSTM cell update ---
#pragma unroll
      for (int i = 0; i < 4; ++i) {
        const int hloc = 16 * wave + 4 * i + quad;              // hidden index
#pragma unroll
        for (int r = 0; r < 8; ++r) {
          // activate own gate once, then broadcast activated values
          float y   = sigmoidf_(s_ * acc[i].f[r]);
          float act = __builtin_fmaf(aa_, y, bb_);
          float ii = quad_bcast(act, qbase + 0);
          float ff = quad_bcast(act, qbase + 1);
          float gz = quad_bcast(act, qbase + 2);
          float oo = quad_bcast(act, qbase + 3);
          float c  = __builtin_fmaf(ff, cst[i][r], ii * gz);
          cst[i][r] = c;
          float hn = oo * tanh_fast(c);
          if (g == 0) {                                         // quad leader
            const int m = r + 8 * half;                         // batch row
            ash[m * KPAD + hloc] = f2bf(hn);                    // raw recurrence
            if (hloc < Hsz) {
              if (layer < Lsz - 1) {
                float dm = dmask[(((size_t)layer * Bsz + (bbase + m)) * Tsz + t) * Hsz + hloc];
                actOut[((size_t)(bbase + m) * Tsz + t) * Hsz + hloc] = f2bf(hn * dm * 2.0f);
              } else if (t == Tsz - 1) {                        // final projection
                atomicAdd(&oacc[m], hn * W_out[hloc]);          // ds_add_f32
              }
            }
          }
        }
      }
      __syncthreads();
    }
    __syncthreads();
  }

  if (tid < BT) out[bbase + tid] = oacc[tid] + b_out[0];
}

// ---------------------------------------------------------------------------
extern "C" void kernel_launch(void* const* d_in, const int* in_sizes, int n_in,
                              void* d_out, int out_size, void* d_ws, size_t ws_size,
                              hipStream_t stream) {
  const float* x         = (const float*)d_in[0];
  const float* W_ih0     = (const float*)d_in[1];
  const float* W_ih_rest = (const float*)d_in[2];
  const float* W_hh      = (const float*)d_in[3];
  const float* b_ih      = (const float*)d_in[4];
  const float* b_hh      = (const float*)d_in[5];
  const float* W_out     = (const float*)d_in[6];
  const float* b_out     = (const float*)d_in[7];
  const float* dmask     = (const float*)d_in[8];
  float* out = (float*)d_out;

  // workspace: packed weights (1 MB, L2-resident) + ping-pong bf16 activations
  unsigned short* wbuf = (unsigned short*)d_ws;
  unsigned short* actA = wbuf + (size_t)Lsz * NP * KPAD;
  unsigned short* actB = actA + (size_t)Bsz * Tsz * Hsz;

  const int totalW = Lsz * NP * KPAD;
  lstm_pack_weights<<<(totalW + 255) / 256, 256, 0, stream>>>(
      W_ih0, W_ih_rest, W_hh, b_ih, b_hh, wbuf);
  lstm_persist<<<Bsz / BT, BLK, 0, stream>>>(x, wbuf, W_out, b_out, dmask,
                                             actA, actB, out);
}
